// SelfAttnLayer_7679401525439
// MI455X (gfx1250) — matrix operand order repaired and checked by reading, not compiled, not run on hardware
//
#include <hip/hip_runtime.h>
#include <hip/hip_bf16.h>

// ---------------------------------------------------------------------------
// SelfAttn: out = gamma * softmax(Q K^T) V + X,  Q=K=X@WQ (8,4096,32),
// V=X@WV (8,4096,256).  Flash-attention style, all matmuls on
// v_wmma_f32_16x16x32_f16; K/V blocks streamed into LDS with
// global_load_async_to_lds_* (ASYNCcnt), double-buffered. (gfx1250, wave32)
// ---------------------------------------------------------------------------

typedef __attribute__((ext_vector_type(16))) _Float16 v16h;
typedef __attribute__((ext_vector_type(8)))  _Float16 v8h;
typedef __attribute__((ext_vector_type(8)))  float    v8f;
typedef __attribute__((ext_vector_type(4)))  float    v4f;

#define NBATCH 8
#define NTOK   4096        // H*W
#define NC     256
#define CQ     32          // NC / k_ratio
#define LOG2E  1.44269504088896340736f

static __device__ __forceinline__ v8f wmma_f16(v16h a, v16h b, v8f c) {
    return __builtin_amdgcn_wmma_f32_16x16x32_f16(false, a, false, b,
                                                  (short)0, c, false, false);
}

static __device__ __forceinline__ v16h ld2_v8h(const _Float16* p) {
    v8h a = *(const v8h*)p;
    v8h b = *(const v8h*)(p + 8);
    v16h r;
#pragma unroll
    for (int i = 0; i < 8; ++i) { r[i] = a[i]; r[8 + i] = b[i]; }
    return r;
}

// ---- CDNA5 async memory->LDS (ASYNCcnt) -----------------------------------
static __device__ __forceinline__ void async_ld_b128(const _Float16* lds_dst,
                                                     const _Float16* gsrc) {
    uint32_t lds = (uint32_t)(uintptr_t)lds_dst;   // flat LDS addr: low 32 = offset
    uint64_t g   = (uint64_t)(uintptr_t)gsrc;
    asm volatile("global_load_async_to_lds_b128 %0, %1, off"
                 :: "v"(lds), "v"(g) : "memory");
}
static __device__ __forceinline__ void async_ld_b64(const _Float16* lds_dst,
                                                    const _Float16* gsrc) {
    uint32_t lds = (uint32_t)(uintptr_t)lds_dst;
    uint64_t g   = (uint64_t)(uintptr_t)gsrc;
    asm volatile("global_load_async_to_lds_b64 %0, %1, off"
                 :: "v"(lds), "v"(g) : "memory");
}
static __device__ __forceinline__ void wait_async0() {
    asm volatile("s_wait_asynccnt 0x0" ::: "memory");
}

// ---------------------------------------------------------------------------
// Kernel 1: convert weights f32 -> f16 (row-major == WMMA-B friendly:
// B fragment is lane = n-col, halves = k-rows).
// ---------------------------------------------------------------------------
__global__ void cvt_weights(const float* __restrict__ WQ,
                            const float* __restrict__ WV,
                            _Float16* __restrict__ WQh,
                            _Float16* __restrict__ WVh) {
    int i = blockIdx.x * blockDim.x + threadIdx.x;   // 0 .. 73727
    if (i < NC * CQ) {
        WQh[i] = (_Float16)WQ[i];
    } else {
        int j = i - NC * CQ;
        if (j < NC * NC) WVh[j] = (_Float16)WV[j];
    }
}

// ---------------------------------------------------------------------------
// Kernel 2: projections via WMMA.
//   rows = flattened (b, token): 32768, tiled by 16 -> blockIdx.x (2048)
//   col tiles: 0..1 -> Q (32 cols), 2..17 -> V (256 cols) -> blockIdx.y (18)
// Outputs: Qh (32768,32) row-major, Qt (8,32,4096) channel-major,
//          Vh (32768,256) row-major.
// ---------------------------------------------------------------------------
__global__ void proj_kernel(const float* __restrict__ X,
                            const _Float16* __restrict__ WQh,
                            const _Float16* __restrict__ WVh,
                            _Float16* __restrict__ Qh,
                            _Float16* __restrict__ Qt,
                            _Float16* __restrict__ Vh) {
    const int tile = blockIdx.x;           // 16-row tile over B*NTOK
    const int ct   = blockIdx.y;           // 0..17
    const int lane = threadIdx.x & 31;
    const int l15  = lane & 15;
    const int hi   = lane >> 4;

    const int row = tile * 16 + l15;       // A-row this lane supplies
    const float* xrow = X + (size_t)row * NC;

    const bool isQ = (ct < 2);
    const _Float16* Wp = isQ ? WQh : WVh;
    const int wcols = isQ ? CQ : NC;
    const int ccol  = isQ ? ct * 16 : (ct - 2) * 16;

    v8f acc = {};
#pragma unroll
    for (int kb = 0; kb < NC / 32; ++kb) {
        // A fragment (f32 -> f16 on the fly):
        //   halves h<8 -> k = hi*8 + h ; h>=8 -> k = 16 + hi*8 + (h-8)
        v4f f0 = *(const v4f*)(xrow + kb * 32 + hi * 8);
        v4f f1 = *(const v4f*)(xrow + kb * 32 + hi * 8 + 4);
        v4f g0 = *(const v4f*)(xrow + kb * 32 + 16 + hi * 8);
        v4f g1 = *(const v4f*)(xrow + kb * 32 + 16 + hi * 8 + 4);
        v16h a;
#pragma unroll
        for (int j = 0; j < 4; ++j) {
            a[j]      = (_Float16)f0[j];
            a[4 + j]  = (_Float16)f1[j];
            a[8 + j]  = (_Float16)g0[j];
            a[12 + j] = (_Float16)g1[j];
        }
        v16h b;
#pragma unroll
        for (int s = 0; s < 16; ++s) {
            const int k = (s < 8) ? (hi * 8 + s) : (16 + hi * 8 + (s - 8));
            b[s] = Wp[(size_t)(kb * 32 + k) * wcols + ccol + l15];
        }
        acc = wmma_f16(a, b, acc);
    }

    // C layout: element (M = r + 8*hi, N = l15) in acc[r]
#pragma unroll
    for (int r = 0; r < 8; ++r) {
        const int trow = tile * 16 + r + hi * 8;
        const _Float16 hv = (_Float16)acc[r];
        if (isQ) {
            const int ch = ct * 16 + l15;
            Qh[(size_t)trow * CQ + ch] = hv;
            const int b   = trow >> 12;
            const int tok = trow & (NTOK - 1);
            Qt[((size_t)b * CQ + ch) * NTOK + tok] = hv;
        } else {
            Vh[(size_t)trow * NC + (ct - 2) * 16 + l15] = hv;
        }
    }
}

// ---------------------------------------------------------------------------
// Kernel 3: fused flash attention.
// Block = 256 threads (8 waves); each wave owns a 16-row q-tile (block covers
// 128 rows of one batch).  Per 32-key block:
//   S^T = K(16x32ch) @ Q^T(32ch x 16rows)   (2 WMMAs)
//   per-lane online softmax (lane = q-row in S^T C-layout);
//   P's A-fragment falls out of S^T C-layout with no shuffles/LDS;
//   O(16x256) += P(16x32) @ V(32x256)        (16 WMMAs).
// K/V blocks double-buffered in LDS via global_load_async_to_lds_*.
// ---------------------------------------------------------------------------
#define VS_STRIDE 264     // 256 + 8 halves pad
#define KS_STRIDE 40      // 32 + 8 halves pad
#define VS_BUF   (32 * VS_STRIDE)
#define KS_BUF   (32 * KS_STRIDE)

__global__ __launch_bounds__(256) void attn_kernel(
        const float* __restrict__ X,
        const _Float16* __restrict__ Qh,
        const _Float16* __restrict__ Qt,
        const _Float16* __restrict__ Vh,
        const float* __restrict__ gamma,
        float* __restrict__ out) {
    __shared__ _Float16 Vs[2 * VS_BUF];   // 2 x (32 keys x 256 ch, padded)
    __shared__ _Float16 Ks[2 * KS_BUF];   // 2 x (32 keys x 32 ch, padded)

    const int b    = blockIdx.y;
    const int wave = threadIdx.x >> 5;
    const int lane = threadIdx.x & 31;
    const int l15  = lane & 15;
    const int hi   = lane >> 4;
    const int rowTile = blockIdx.x * 8 + wave;        // 0..255 within batch

    const _Float16* Qbase = Qh + (size_t)b * NTOK * CQ;
    const _Float16* Vbase = Vh + (size_t)b * NTOK * NC;

    // staging assignment for this thread (fixed across iterations)
    const int skey = threadIdx.x >> 3;          // 0..31
    const int schv = (threadIdx.x & 7) * 32;    // V: 32 halves per thread
    const int schk = (threadIdx.x & 7) * 4;     // K: 4 halves per thread

    // B operand of S^T: Q^T fragment, lane = q-row l15, halves = channels (table k order)
    v16h qb = ld2_v8h(Qbase + (size_t)(rowTile * 16 + l15) * CQ + hi * 8);
    {
        v8h tq = *(const v8h*)(Qbase + (size_t)(rowTile * 16 + l15) * CQ + 16 + hi * 8);
#pragma unroll
        for (int i = 0; i < 8; ++i) qb[8 + i] = tq[i];
    }

    v8f o[16];
#pragma unroll
    for (int ct = 0; ct < 16; ++ct) o[ct] = (v8f){};
    float mrow = -3.0e38f;    // running max, row = l15 (dup across hi halves)
    float lrow = 0.0f;        // running denom

    // prologue: async-stage key block 0 into buffer 0
    {
        _Float16* vdst = Vs + skey * VS_STRIDE + schv;
        const _Float16* vsrc = Vbase + (size_t)skey * NC + schv;
        async_ld_b128(vdst,      vsrc);
        async_ld_b128(vdst + 8,  vsrc + 8);
        async_ld_b128(vdst + 16, vsrc + 16);
        async_ld_b128(vdst + 24, vsrc + 24);
        async_ld_b64(Ks + skey * KS_STRIDE + schk,
                     Qbase + (size_t)skey * CQ + schk);
    }

    int bufsel = 0;
    for (int kb = 0; kb < NTOK; kb += 32) {
        wait_async0();        // my async writes into buffer `bufsel` done
        __syncthreads();      // all waves' writes visible; prev reads done

        if (kb + 32 < NTOK) { // prefetch next block into the other buffer
            const int nb = bufsel ^ 1;
            _Float16* vdst = Vs + nb * VS_BUF + skey * VS_STRIDE + schv;
            const _Float16* vsrc = Vbase + (size_t)(kb + 32 + skey) * NC + schv;
            async_ld_b128(vdst,      vsrc);
            async_ld_b128(vdst + 8,  vsrc + 8);
            async_ld_b128(vdst + 16, vsrc + 16);
            async_ld_b128(vdst + 24, vsrc + 24);
            async_ld_b64(Ks + nb * KS_BUF + skey * KS_STRIDE + schk,
                         Qbase + (size_t)(kb + 32 + skey) * CQ + schk);
        }

        const _Float16* Vcur = Vs + bufsel * VS_BUF;
        const _Float16* Kcur = Ks + bufsel * KS_BUF;

        // ---- S^T = K @ Q^T : A = K rows (lane = key), B = qb ----
        v8f s0 = {}, s1 = {};
        {
            v16h ka0 = ld2_v8h(Kcur + (size_t)(l15) * KS_STRIDE + hi * 8);
            v8h  t0  = *(const v8h*)(Kcur + (size_t)(l15) * KS_STRIDE + 16 + hi * 8);
#pragma unroll
            for (int i = 0; i < 8; ++i) ka0[8 + i] = t0[i];
            s0 = wmma_f16(ka0, qb, s0);

            v16h ka1 = ld2_v8h(Kcur + (size_t)(16 + l15) * KS_STRIDE + hi * 8);
            v8h  t1  = *(const v8h*)(Kcur + (size_t)(16 + l15) * KS_STRIDE + 16 + hi * 8);
#pragma unroll
            for (int i = 0; i < 8; ++i) ka1[8 + i] = t1[i];
            s1 = wmma_f16(ka1, qb, s1);
        }
        // S^T C-layout: lane n = q-row l15, VGPR r = key (r + 8*hi) [+16 tile1]

        // ---- online softmax (per-lane: this lane's q-row) ----
        float mnew = mrow;
#pragma unroll
        for (int r = 0; r < 8; ++r) {
            mnew = fmaxf(mnew, s0[r]);
            mnew = fmaxf(mnew, s1[r]);
        }
        mnew = fmaxf(mnew, __shfl_xor(mnew, 16, 32));
        const float scale = __builtin_exp2f((mrow - mnew) * LOG2E);
        mrow = mnew;

        float psum = 0.0f;
        v16h pa;   // A-fragment of P: h<8 -> tile0 r=h ; h>=8 -> tile1 r=h-8
#pragma unroll
        for (int r = 0; r < 8; ++r) {
            const float e0 = __builtin_exp2f((s0[r] - mrow) * LOG2E);
            const float e1 = __builtin_exp2f((s1[r] - mrow) * LOG2E);
            psum += e0 + e1;
            pa[r]     = (_Float16)e0;
            pa[8 + r] = (_Float16)e1;
        }
        psum += __shfl_xor(psum, 16, 32);
        lrow = lrow * scale + psum;

        // ---- rescale O: O rows are r + 8*hi; stats live at lane = row ----
        float sc[8];
#pragma unroll
        for (int r = 0; r < 8; ++r) sc[r] = __shfl(scale, r + hi * 8, 32);
#pragma unroll
        for (int ct = 0; ct < 16; ++ct)
#pragma unroll
            for (int r = 0; r < 8; ++r) o[ct][r] *= sc[r];

        // ---- O += P @ V : B fragment lane = channel l15, halves = keys (table k order) ----
#pragma unroll
        for (int ct = 0; ct < 16; ++ct) {
            v16h vb;
#pragma unroll
            for (int s = 0; s < 16; ++s) {
                const int k = (s < 8) ? (hi * 8 + s) : (16 + hi * 8 + (s - 8));
                vb[s] = Vcur[(size_t)k * VS_STRIDE + ct * 16 + l15];
            }
            o[ct] = wmma_f16(pa, vb, o[ct]);
        }

        bufsel ^= 1;
    }

    // ---- epilogue: out = gamma * O / l + X ----
    const float inv = 1.0f / lrow;
    float invr[8];
#pragma unroll
    for (int r = 0; r < 8; ++r) invr[r] = __shfl(inv, r + hi * 8, 32);
    const float g = gamma[0];
    const float* xres = X   + ((size_t)b * NTOK + rowTile * 16) * NC;
    float*       obas = out + ((size_t)b * NTOK + rowTile * 16) * NC;
#pragma unroll
    for (int ct = 0; ct < 16; ++ct) {
#pragma unroll
        for (int r = 0; r < 8; ++r) {
            const int m  = r + 8 * hi;
            const int ch = ct * 16 + l15;
            obas[(size_t)m * NC + ch] =
                g * (o[ct][r] * invr[r]) + xres[(size_t)m * NC + ch];
        }
    }
}

// ---------------------------------------------------------------------------
extern "C" void kernel_launch(void* const* d_in, const int* in_sizes, int n_in,
                              void* d_out, int out_size, void* d_ws, size_t ws_size,
                              hipStream_t stream) {
    const float* X     = (const float*)d_in[0];   // (8,64,64,256)
    const float* WQ    = (const float*)d_in[1];   // (256,32)
    const float* WV    = (const float*)d_in[2];   // (256,256)
    const float* gamma = (const float*)d_in[3];   // (1,)

    char* ws = (char*)d_ws;
    _Float16* WQh = (_Float16*)(ws);                         //    16 KB
    _Float16* WVh = (_Float16*)(ws + 16384);                 //   128 KB
    _Float16* Qh  = (_Float16*)(ws + 147456);                //     2 MB
    _Float16* Qt  = (_Float16*)(ws + 2244608);               //     2 MB
    _Float16* Vh  = (_Float16*)(ws + 4341760);               //    16 MB
    (void)ws_size; (void)in_sizes; (void)n_in; (void)out_size;

    cvt_weights<<<288, 256, 0, stream>>>(WQ, WV, WQh, WVh);
    proj_kernel<<<dim3(2048, 18), 32, 0, stream>>>(X, WQh, WVh, Qh, Qt, Vh);
    attn_kernel<<<dim3(32, 8), 256, 0, stream>>>(X, Qh, Qt, Vh, gamma,
                                                 (float*)d_out);
}
